// MultiHeadAttention_20109036880334
// MI455X (gfx1250) — matrix operand order repaired and checked by reading, not compiled, not run on hardware
//
#include <hip/hip_runtime.h>
#include <hip/hip_bf16.h>

// ---------------- problem constants ----------------
constexpr int Bc = 64, Lc = 52, DM = 4096, Hc = 32, DK = 128, DV = 128, Cc = 128;
constexpr int LP = 64;                 // L padded to multiple of 16
constexpr int BH = Bc * Hc;            // 2048
constexpr int Mrows = Bc * Lc;         // 3328 = 208*16

typedef __attribute__((ext_vector_type(16))) __bf16 v16bf;
typedef __attribute__((ext_vector_type(8)))  __bf16 v8bf;
typedef __attribute__((ext_vector_type(8)))  float  v8f;
typedef int v4i __attribute__((vector_size(16)));   // matches async-LDS builtin param

__device__ __forceinline__ v16bf cat8(v8bf lo, v8bf hi) {
  return __builtin_shufflevector(lo, hi, 0,1,2,3,4,5,6,7,8,9,10,11,12,13,14,15);
}
// A-fragment (16x32 bf16): lane M = lane&15; elems 0..7 -> K = kc*32 + kg*8 + e,
// elems 8..15 -> K = kc*32 + 16 + kg*8 + (e-8). Two contiguous 16B loads.
__device__ __forceinline__ v16bf load_afrag(const __bf16* row, int kc, int kg) {
  const __bf16* p = row + kc * 32 + kg * 8;
  return cat8(*(const v8bf*)p, *(const v8bf*)(p + 16));
}
// B-fragment (32x16 bf16) from B^T row-major: lane N = lane&15;
// elems 0..7 -> K = kc*32 + kg*8 + e, elems 8..15 -> K = kc*32 + 16 + kg*8 + (e-8).
__device__ __forceinline__ v16bf load_bfrag(const __bf16* rowT, int kc, int kg) {
  const __bf16* p = rowT + kc * 32 + kg * 8;
  return cat8(*(const v8bf*)p, *(const v8bf*)(p + 16));
}
#define WMMA_BF16(A, Bm, Cm) \
  __builtin_amdgcn_wmma_f32_16x16x32_bf16(false, (A), false, (Bm), (short)0, (Cm), false, false)

// ---- CDNA5 async global->LDS copy (ASYNCcnt-tracked), with sync fallback ----
#if defined(__has_builtin)
#if __has_builtin(__builtin_amdgcn_global_load_async_to_lds_b128) && \
    __has_builtin(__builtin_amdgcn_s_wait_asynccnt)
#define HAVE_ASYNC_LDS 1
#endif
#endif

__device__ __forceinline__ void async_cp16(const __bf16* g, __bf16* l) {
#ifdef HAVE_ASYNC_LDS
  __builtin_amdgcn_global_load_async_to_lds_b128(
      (__attribute__((address_space(1))) v4i*)(const void*)g,
      (__attribute__((address_space(3))) v4i*)(void*)l, 0, 0);
#else
  *(v8bf*)l = *(const v8bf*)g;
#endif
}
__device__ __forceinline__ void wait_async_all() {
#ifdef HAVE_ASYNC_LDS
  __builtin_amdgcn_s_wait_asynccnt(0);
#endif
}

// ---------------- kernel 1: per-head weight transpose f32 -> bf16 ----------------
// in:  W[h][c][k]  (H,128,128) f32     out: WT[h][k][c] bf16
__global__ __launch_bounds__(256) void transpose_w_kernel(
    const float* __restrict__ Wq, const float* __restrict__ Wk, const float* __restrict__ Wv,
    __bf16* __restrict__ WqT, __bf16* __restrict__ WkT, __bf16* __restrict__ WvT) {
  const float* in  = (blockIdx.z == 0) ? Wq : (blockIdx.z == 1) ? Wk : Wv;
  __bf16*      out = (blockIdx.z == 0) ? WqT : (blockIdx.z == 1) ? WkT : WvT;
  int idx = blockIdx.x * 256 + threadIdx.x;       // H*128*128 = 524288 elements
  int h = idx >> 14, rem = idx & 16383;
  int k = rem >> 7, c = rem & 127;
  out[idx] = (__bf16)in[(h << 14) + (c << 7) + k];
}

// ---------------- kernel 2: Wo transpose f32 -> bf16 ----------------
// in: Wo[k][n] (4096,4096)   out: WoT[n][k]
__global__ __launch_bounds__(256) void transpose_wo_kernel(
    const float* __restrict__ Wo, __bf16* __restrict__ WoT) {
  size_t idx = (size_t)blockIdx.x * 256 + threadIdx.x;   // 16.78M
  int n = (int)(idx >> 12), k = (int)(idx & 4095);
  WoT[idx] = (__bf16)Wo[(size_t)k * DM + n];
}

// ---------------- kernel 3: q/k/v projection ----------------
// One wave per (b*H+h, mtile, which). A = raw-reshape slice (52x128 f32, zero-padded
// to 64 rows), B^T = WT[h] (128x128 bf16). q,k stored row-major [LP x DK] bf16;
// v stored transposed [DV x LP] bf16.
__global__ __launch_bounds__(32) void proj_kernel(
    const float* __restrict__ Qf, const float* __restrict__ Kf, const float* __restrict__ Vf,
    const __bf16* __restrict__ WqT, const __bf16* __restrict__ WkT, const __bf16* __restrict__ WvT,
    const float* __restrict__ bq, const float* __restrict__ bk, const float* __restrict__ bv,
    __bf16* __restrict__ qb, __bf16* __restrict__ kb, __bf16* __restrict__ vtb) {
  const int z = blockIdx.z;
  const float*  X    = (z == 0) ? Qf : (z == 1) ? Kf : Vf;
  const __bf16* WT   = (z == 0) ? WqT : (z == 1) ? WkT : WvT;
  const float*  bias = (z == 0) ? bq : (z == 1) ? bk : bv;

  const int bh = blockIdx.x;                 // b*H + h
  const int h  = bh & (Hc - 1);
  const int m0 = blockIdx.y * 16;
  const int lane = threadIdx.x, n15 = lane & 15, kg = lane >> 4;
  const int M = m0 + n15;                    // A row (padded space)
  const bool rowValid = (M < Lc);
  // raw reshape: split(X)[b,h,l,c] = X[((b*H + h)*L + l)*C + c]
  const float* xrow = X + ((size_t)bh * Lc + M) * Cc;
  const __bf16* WTh = WT + (size_t)h * DK * Cc;

  v8f acc[8] = {};
#pragma unroll
  for (int kc = 0; kc < 4; ++kc) {
    v16bf a = {};
    if (rowValid) {
      const float* p0 = xrow + kc * 32 + kg * 8;
      const float* p1 = p0 + 16;
      float4 r0 = *(const float4*)p0, r1 = *(const float4*)(p0 + 4);
      float4 r2 = *(const float4*)p1, r3 = *(const float4*)(p1 + 4);
      a[0]=(__bf16)r0.x; a[1]=(__bf16)r0.y; a[2]=(__bf16)r0.z; a[3]=(__bf16)r0.w;
      a[4]=(__bf16)r1.x; a[5]=(__bf16)r1.y; a[6]=(__bf16)r1.z; a[7]=(__bf16)r1.w;
      a[8]=(__bf16)r2.x; a[9]=(__bf16)r2.y; a[10]=(__bf16)r2.z; a[11]=(__bf16)r2.w;
      a[12]=(__bf16)r3.x; a[13]=(__bf16)r3.y; a[14]=(__bf16)r3.z; a[15]=(__bf16)r3.w;
    }
#pragma unroll
    for (int n = 0; n < 8; ++n) {
      v16bf bfrag = load_bfrag(WTh + (size_t)(n * 16 + n15) * Cc, kc, kg);
      acc[n] = WMMA_BF16(a, bfrag, acc[n]);
    }
  }

#pragma unroll
  for (int n = 0; n < 8; ++n) {
    const int col = n * 16 + n15;
    const float bvl = bias[h * DK + col];
    if (z != 2) {
      __bf16* out = (z == 0) ? qb : kb;
#pragma unroll
      for (int v = 0; v < 8; ++v) {
        int row = m0 + kg * 8 + v;
        out[((size_t)bh * LP + row) * DK + col] = (__bf16)(acc[n][v] + bvl);
      }
    } else {
      v8bf pack;
#pragma unroll
      for (int v = 0; v < 8; ++v) pack[v] = (__bf16)(acc[n][v] + bvl);
      // vT[bh][d][l] with l = m0 + kg*8 + v contiguous
      *(v8bf*)(vtb + ((size_t)bh * DV + col) * LP + m0 + kg * 8) = pack;
    }
  }
}

// ---------------- kernel 4: per-head causal attention ----------------
__global__ __launch_bounds__(256) void attn_kernel(
    const __bf16* __restrict__ qb, const __bf16* __restrict__ kb,
    const __bf16* __restrict__ vtb, __bf16* __restrict__ ob) {
  __shared__ float  S[64][64];
  __shared__ __bf16 P[64][64];
  const int bh = blockIdx.x;
  const int b = bh >> 5, h = bh & 31;
  const int tid = threadIdx.x, wave = tid >> 5, lane = tid & 31;
  const int n15 = lane & 15, kg = lane >> 4;
  const __bf16* qh = qb  + (size_t)bh * LP * DK;
  const __bf16* kh = kb  + (size_t)bh * LP * DK;
  const __bf16* vh = vtb + (size_t)bh * DV * LP;
  const float scale = 0.08838834764831845f;       // 1/sqrt(128)
  const float NEGINF = -__builtin_inff();

  // --- S = (q k^T) * scale, causal-masked, into LDS (16 tiles / 8 waves) ---
  for (int t = wave; t < 16; t += 8) {
    const int mt = t >> 2, nt = t & 3;
    v8f acc = {};
#pragma unroll
    for (int kc = 0; kc < 4; ++kc) {
      v16bf a  = load_afrag(qh + (size_t)(mt * 16 + n15) * DK, kc, kg);
      v16bf bb = load_bfrag(kh + (size_t)(nt * 16 + n15) * DK, kc, kg);
      acc = WMMA_BF16(a, bb, acc);
    }
#pragma unroll
    for (int v = 0; v < 8; ++v) {
      const int i = mt * 16 + kg * 8 + v, j = nt * 16 + n15;
      S[i][j] = (j <= i) ? acc[v] * scale : NEGINF;
    }
  }
  __syncthreads();

  // --- row softmax (one thread per row) ---
  if (tid < 64) {
    const int r = tid;
    float m = NEGINF;
    for (int j = 0; j < 64; ++j) m = fmaxf(m, S[r][j]);
    float s = 0.f;
    for (int j = 0; j < 64; ++j) { float e = __expf(S[r][j] - m); S[r][j] = e; s += e; }
    const float inv = 1.f / s;
    for (int j = 0; j < 64; ++j) P[r][j] = (__bf16)(S[r][j] * inv);
  }
  __syncthreads();

  // --- O = P @ v  (32 tiles / 8 waves), store concat-heads bf16 ---
  for (int t = wave; t < 32; t += 8) {
    const int mt = t >> 3, nt = t & 7;
    v8f acc = {};
#pragma unroll
    for (int kc = 0; kc < 2; ++kc) {
      v16bf a  = load_afrag(&P[mt * 16 + n15][0], kc, kg);
      v16bf bb = load_bfrag(vh + (size_t)(nt * 16 + n15) * LP, kc, kg);
      acc = WMMA_BF16(a, bb, acc);
    }
    const int d = nt * 16 + n15;
#pragma unroll
    for (int v = 0; v < 8; ++v) {
      const int i = mt * 16 + kg * 8 + v;
      if (i < Lc)
        ob[((size_t)(b * Lc + i)) * DM + h * DV + d] = (__bf16)acc[v];
    }
  }
}

// ---------------- kernel 5: output projection (3328x4096)@(4096x4096)+bo ----------------
// 8 waves/block share one 16-row A strip: staged in LDS (double-buffered 2x16KB),
// filled with async global->LDS copies overlapped with the WMMA on the previous slab.
__global__ __launch_bounds__(256) void out_gemm_kernel(
    const __bf16* __restrict__ ob, const __bf16* __restrict__ WoT,
    const float* __restrict__ bo, float* __restrict__ out) {
  __shared__ __bf16 Alds[2][16][512];               // 2 x 16 KB
  const int mt = blockIdx.y;                        // 0..207
  const int tid = threadIdx.x;
  const int wave = tid >> 5, lane = tid & 31;
  const int n15 = lane & 15, kg = lane >> 4;
  const int nbase = blockIdx.x * 1024 + wave * 128; // 8 N-tiles per wave
  const __bf16* Ablk = ob + (size_t)(mt * 16) * DM;

  // stage k-slab s (16 rows x 512 cols bf16 = 16 KB) into buffer buf
  auto stage = [&](int s, int buf) {
    const __bf16* src = Ablk + s * 512;
#pragma unroll
    for (int i = 0; i < 4; ++i) {
      int c = tid + 256 * i;        // 16B-chunk id, 0..1023 (64 chunks/row)
      int row = c >> 6;
      int col = (c & 63) * 8;       // element offset (8 bf16 per 16B)
      async_cp16(src + (size_t)row * DM + col, &Alds[buf][row][col]);
    }
  };

  v8f acc[8] = {};
  stage(0, 0);
  for (int s = 0; s < 8; ++s) {
    const int buf = s & 1;
    wait_async_all();               // this slab's copies complete (ASYNCcnt==0)
    __syncthreads();                // visible to all 8 waves
    if (s + 1 < 8) stage(s + 1, buf ^ 1);   // overlap next copy with compute
    const __bf16* arow = &Alds[buf][n15][0];
    for (int kcl = 0; kcl < 16; ++kcl) {
      v16bf a = load_afrag(arow, kcl, kg);
#pragma unroll
      for (int n = 0; n < 8; ++n) {
        v16bf bb = load_bfrag(WoT + (size_t)(nbase + n * 16 + n15) * DM + s * 512, kcl, kg);
        acc[n] = WMMA_BF16(a, bb, acc[n]);
      }
    }
    __syncthreads();                // all waves done reading buf before reuse
  }

#pragma unroll
  for (int n = 0; n < 8; ++n) {
    const int col = nbase + n * 16 + n15;
    const float bvl = bo[col];
#pragma unroll
    for (int v = 0; v < 8; ++v) {
      const int row = mt * 16 + kg * 8 + v;
      out[(size_t)row * DM + col] = acc[n][v] + bvl;
    }
  }
}

// ---------------- host launcher ----------------
extern "C" void kernel_launch(void* const* d_in, const int* in_sizes, int n_in,
                              void* d_out, int out_size, void* d_ws, size_t ws_size,
                              hipStream_t stream) {
  (void)in_sizes; (void)n_in; (void)out_size; (void)ws_size;
  const float* Q  = (const float*)d_in[0];
  const float* K  = (const float*)d_in[1];
  const float* V  = (const float*)d_in[2];
  const float* Wq = (const float*)d_in[3];
  const float* bq = (const float*)d_in[4];
  const float* Wk = (const float*)d_in[5];
  const float* bk = (const float*)d_in[6];
  const float* Wv = (const float*)d_in[7];
  const float* bv = (const float*)d_in[8];
  const float* Wo = (const float*)d_in[9];
  const float* bo = (const float*)d_in[10];
  float* out = (float*)d_out;

  char* ws = (char*)d_ws;
  size_t off = 0;
  auto take = [&](size_t bytes) -> char* {
    char* p = ws + off;
    off += (bytes + 255) & ~(size_t)255;
    return p;
  };
  __bf16* WqT = (__bf16*)take((size_t)Hc * DK * Cc * 2);          // 1 MB
  __bf16* WkT = (__bf16*)take((size_t)Hc * DK * Cc * 2);          // 1 MB
  __bf16* WvT = (__bf16*)take((size_t)Hc * DV * Cc * 2);          // 1 MB
  __bf16* WoT = (__bf16*)take((size_t)DM * DM * 2);               // 32 MB
  __bf16* qb  = (__bf16*)take((size_t)BH * LP * DK * 2);          // 32 MB
  __bf16* kb  = (__bf16*)take((size_t)BH * LP * DK * 2);          // 32 MB
  __bf16* vtb = (__bf16*)take((size_t)BH * DV * LP * 2);          // 32 MB
  __bf16* ob  = (__bf16*)take((size_t)Mrows * DM * 2);            // 27.25 MB

  transpose_w_kernel<<<dim3(Hc * DK * Cc / 256, 1, 3), 256, 0, stream>>>(
      Wq, Wk, Wv, WqT, WkT, WvT);
  transpose_wo_kernel<<<dim3((unsigned)((size_t)DM * DM / 256)), 256, 0, stream>>>(Wo, WoT);
  proj_kernel<<<dim3(BH, LP / 16, 3), 32, 0, stream>>>(
      Q, K, V, WqT, WkT, WvT, bq, bk, bv, qb, kb, vtb);
  attn_kernel<<<dim3(BH), 256, 0, stream>>>(qb, kb, vtb, ob);
  out_gemm_kernel<<<dim3(4, Mrows / 16), 256, 0, stream>>>(ob, WoT, bo, out);
}
